// Encoder_68178310857143
// MI455X (gfx1250) — matrix operand order repaired and checked by reading, not compiled, not run on hardware
//
#include <hip/hip_runtime.h>
#include <hip/hip_bf16.h>

// ---------------------------------------------------------------------------
// MI455X (gfx1250): implicit-GEMM f16 WMMA convolutions, v3.
//  - activations materialized once per layer as normalized+ReLU f16
//    (halves the L2 gather traffic; gather becomes a pure f16 copy)
//  - double-buffered LDS A tile: one barrier per K-chunk, staging overlaps WMMA
//  - 4 v_wmma_f32_16x16x32_f16 per wave per K-chunk (64px x 16co)
//  - weights pre-packed per layer into exact B-fragment order (f16, L2-resident)
//  - working set: 134MB f32 raw + 2x67MB f16 ping-pong -> L2 (192MB) friendly
//  - instance pooling: LDS ds_add_f32 partials -> few global_atomic_add_f32
// ---------------------------------------------------------------------------

typedef __attribute__((ext_vector_type(16))) _Float16 v16h;
typedef __attribute__((ext_vector_type(4)))  _Float16 v4h;
typedef __attribute__((ext_vector_type(8)))  float    v8f;

// Slot of k (0..31) in an A row (contiguous order, matching packed B):  q = k
__device__ __forceinline__ int a_perm(int k) {
    return k;
}

#define A_STRIDE 40   // halfs per A row: 80B, 16B-aligned fragments, bank-spread

// ---------------------------------------------------------------------------
// Per-layer weight pack -> f16 B-fragment order:
//   wp[(((nt*NC + c)*32 + lane)*16 + e]
// ---------------------------------------------------------------------------
__launch_bounds__(32)
__global__ void pack_weights(const float* __restrict__ w, _Float16* __restrict__ wp,
                             int Cin, int Cout, int KSZ, int trans) {
    const int KK   = KSZ * KSZ;
    const int Ktot = Cin * KK;
    const int lane = threadIdx.x;
    const int c    = blockIdx.x;          // K-chunk
    const int nt   = blockIdx.y;          // cout tile
    const int n    = lane & 15;
    const int kb   = (lane < 16) ? 0 : 16;
    const int co   = nt * 16 + n;

    v16h f{};
#pragma unroll
    for (int vv = 0; vv < 8; ++vv) {
#pragma unroll
        for (int j = 0; j < 2; ++j) {
            const int k = c * 32 + kb + 2 * vv + j;
            float v = 0.0f;
            if (k < Ktot && co < Cout) {
                const int ci = k / KK;
                const int r  = k - ci * KK;
                const int kh = r / KSZ;
                const int kw = r - kh * KSZ;
                long wi;
                if (trans)   // torch (Cin,Cout,kh,kw), spatially flipped
                    wi = ((long)(ci * Cout + co) * KSZ + (KSZ - 1 - kh)) * KSZ
                         + (KSZ - 1 - kw);
                else         // OIHW
                    wi = ((long)(co * Cin + ci) * KSZ + kh) * KSZ + kw;
                v = w[wi];
            }
            f[2 * vv + j] = (_Float16)v;
        }
    }
    *(v16h*)(wp + (((long)nt * gridDim.x + c) * 32 + lane) * 16) = f;
}

// ---------------------------------------------------------------------------
// (x - mean) * rsqrt(var+eps), optional ReLU, f32 -> f16. Also used as a plain
// f32->f16 convert for the network input (mean == nullptr).
// ---------------------------------------------------------------------------
__launch_bounds__(256)
__global__ void norm_to_f16(const float* __restrict__ x, const float* __restrict__ mean,
                            const float* __restrict__ inv, _Float16* __restrict__ o,
                            int P, int relu) {
    const int bc = blockIdx.y;
    float mu = 0.0f, iv = 1.0f;
    if (mean) { mu = mean[bc]; iv = inv[bc]; }
    const float4* src = (const float4*)(x + (long)bc * P);
    v4h* dst = (v4h*)(o + (long)bc * P);
    const int n4 = P >> 2;
    for (int i = blockIdx.x * 256 + threadIdx.x; i < n4; i += gridDim.x * 256) {
        const float4 v = src[i];
        float a0 = (v.x - mu) * iv;
        float a1 = (v.y - mu) * iv;
        float a2 = (v.z - mu) * iv;
        float a3 = (v.w - mu) * iv;
        if (relu) {
            a0 = a0 > 0.0f ? a0 : 0.0f;
            a1 = a1 > 0.0f ? a1 : 0.0f;
            a2 = a2 > 0.0f ? a2 : 0.0f;
            a3 = a3 > 0.0f ? a3 : 0.0f;
        }
        v4h h;
        h[0] = (_Float16)a0; h[1] = (_Float16)a1;
        h[2] = (_Float16)a2; h[3] = (_Float16)a3;
        dst[i] = h;
    }
}

// ---------------------------------------------------------------------------
// Implicit-GEMM conv over f16 activations. Block = 128 threads (4 waves),
// 64-pixel x 64-cout macro tile, double-buffered LDS A staging.
// ---------------------------------------------------------------------------
template <int KSZ, bool TRANS, bool REFLECT, bool TANH>
__launch_bounds__(128)
__global__ void conv_wmma(const _Float16* __restrict__ in, const _Float16* __restrict__ wp,
                          const float* __restrict__ bias, float* __restrict__ out,
                          int Cin, int lwin, int Cout, int lwout,
                          int stride, int pad, int nchunks) {
    constexpr int KK = KSZ * KSZ;
    const int tid    = threadIdx.x;
    const int lane   = tid & 31;
    const int waveid = tid >> 5;
    const int b      = blockIdx.z;
    const int Hin    = 1 << lwin;
    const int Win    = Hin;
    const int Wout   = 1 << lwout;
    const int P      = Wout << lwout;
    const int Ktot   = Cin * KK;
    const int ntiles = (Cout + 15) >> 4;
    const int nt     = blockIdx.y * 4 + waveid;
    const bool active = (nt < ntiles);
    const int mbase  = blockIdx.x * 64;

    __shared__ __align__(32) _Float16 lA[2][64 * A_STRIDE];

    v8f acc[4] = {{}, {}, {}, {}};
    const int hi   = (lane >= 16) ? 1 : 0;
    const int lmod = lane & 15;

    // This thread's staging role: fixed k-within-chunk, 16 pixel rows.
    const int kk = tid & 31;
    const int mg = tid >> 5;
    const int q  = a_perm(kk);

    auto stage = [&](int c, _Float16* dst) {
        const int k    = c * 32 + kk;
        const bool kval = (k < Ktot);
        const int ci   = kval ? (k / KK) : 0;          // constant-divisor div
        const int r    = k - ci * KK;
        const int kh   = r / KSZ;
        const int kw   = r - kh * KSZ;
        const _Float16* __restrict__ src =
            in + ((long)(b * Cin + ci) << (lwin + lwin));
#pragma unroll 4
        for (int i = 0; i < 16; ++i) {
            const int m  = mg + i * 4;
            const int p  = mbase + m;
            const int oh = p >> lwout;
            const int ow = p & (Wout - 1);
            _Float16 v = (_Float16)0.0f;
            if (kval) {
                int ih, iw;
                bool ok = true;
                if (TRANS) {
                    const int r2 = oh - 1 + kh;
                    const int c2 = ow - 1 + kw;
                    ok = (r2 >= 0) && (c2 >= 0) && ((r2 & 1) == 0) && ((c2 & 1) == 0);
                    ih = r2 >> 1;
                    iw = c2 >> 1;
                    ok = ok && (ih < Hin) && (iw < Win);
                } else {
                    ih = oh * stride - pad + kh;
                    iw = ow * stride - pad + kw;
                    if (REFLECT) {
                        ih = (ih < 0) ? -ih : ((ih >= Hin) ? 2 * Hin - 2 - ih : ih);
                        iw = (iw < 0) ? -iw : ((iw >= Win) ? 2 * Win - 2 - iw : iw);
                    } else {
                        ok = (ih >= 0) && (ih < Hin) && (iw >= 0) && (iw < Win);
                    }
                }
                if (ok) v = src[((long)ih << lwin) + iw];
            }
            dst[m * A_STRIDE + q] = v;
        }
    };

    stage(0, lA[0]);
    __syncthreads();

    for (int c = 0; c < nchunks; ++c) {
        const _Float16* cur = lA[c & 1];
        // B fragment for this chunk (prepacked, contiguous 32B per lane).
        v16h fb{};
        if (active) {
            const _Float16* wpt = wp + (((long)nt * nchunks + c) * 32 + lane) * 16;
            fb = *(const v16h*)wpt;
            __builtin_prefetch(wpt + 512, 0, 1);   // next chunk's B tile
        }
        // Overlap: stage next chunk into the other buffer while doing WMMA.
        if (c + 1 < nchunks) stage(c + 1, lA[(c + 1) & 1]);
        if (active) {
#pragma unroll
            for (int t4 = 0; t4 < 4; ++t4) {
                const v16h fa =
                    *(const v16h*)&cur[(t4 * 16 + lmod) * A_STRIDE + hi * 16];
                acc[t4] = __builtin_amdgcn_wmma_f32_16x16x32_f16(
                    false, fa, false, fb, (short)0, acc[t4], false, false);
            }
        }
        __syncthreads();
    }

    // Writeout: lane = cout, acc rows = consecutive pixels; float4 x2 stores.
    if (active) {
        const int co = nt * 16 + lmod;
        if (co < Cout) {
            const float bs = bias[co];
            float* o = out + (long)(b * Cout + co) * P + mbase + hi * 8;
#pragma unroll
            for (int t4 = 0; t4 < 4; ++t4) {
                float4 lo, hi4;
                lo.x = acc[t4][0] + bs; lo.y = acc[t4][1] + bs;
                lo.z = acc[t4][2] + bs; lo.w = acc[t4][3] + bs;
                hi4.x = acc[t4][4] + bs; hi4.y = acc[t4][5] + bs;
                hi4.z = acc[t4][6] + bs; hi4.w = acc[t4][7] + bs;
                if (TANH) {
                    lo.x = tanhf(lo.x); lo.y = tanhf(lo.y);
                    lo.z = tanhf(lo.z); lo.w = tanhf(lo.w);
                    hi4.x = tanhf(hi4.x); hi4.y = tanhf(hi4.y);
                    hi4.z = tanhf(hi4.z); hi4.w = tanhf(hi4.w);
                }
                *(float4*)(o + t4 * 16)     = lo;
                *(float4*)(o + t4 * 16 + 4) = hi4;
            }
        }
    }
}

// Per-(b,c) mean and rsqrt(biased var + eps); float4 loads.
__launch_bounds__(256)
__global__ void inorm_stats(const float* __restrict__ x, float* mean, float* inv, int P) {
    const int bc = blockIdx.x;
    const float4* p4 = (const float4*)(x + (long)bc * P);
    const int n4 = P >> 2;
    float s = 0.0f, s2 = 0.0f;
    for (int i = threadIdx.x; i < n4; i += 256) {
        const float4 v = p4[i];
        s  += v.x + v.y + v.z + v.w;
        s2 += v.x * v.x + v.y * v.y + v.z * v.z + v.w * v.w;
    }
    __shared__ float sh[256], sh2[256];
    sh[threadIdx.x] = s; sh2[threadIdx.x] = s2;
    __syncthreads();
    for (int o = 128; o > 0; o >>= 1) {
        if (threadIdx.x < o) {
            sh[threadIdx.x]  += sh[threadIdx.x + o];
            sh2[threadIdx.x] += sh2[threadIdx.x + o];
        }
        __syncthreads();
    }
    if (threadIdx.x == 0) {
        const float m   = sh[0] / (float)P;
        const float var = sh2[0] / (float)P - m * m;
        mean[bc] = m;
        inv[bc]  = rsqrtf(var + 1e-5f);
    }
}

__global__ void pool_zero(float* p) {
    const int i = blockIdx.x * 256 + threadIdx.x;
    if (i < 1024) p[i] = 0.0f;
}

__launch_bounds__(256)
__global__ void pool_accum(const float* __restrict__ y, const int* __restrict__ ids,
                           float* sums, float* cnt) {
    const int b = blockIdx.y;
    const int P = 512 * 512;
    __shared__ float ls[32 * 3];
    __shared__ float lc[32];
    for (int i = threadIdx.x; i < 96; i += 256) ls[i] = 0.0f;
    for (int i = threadIdx.x; i < 32; i += 256) lc[i] = 0.0f;
    __syncthreads();
    const int p = blockIdx.x * 256 + threadIdx.x;
    if (p < P) {
        const int id = ids[(long)b * P + p] & 31;
        atomicAdd(&lc[id], 1.0f);
#pragma unroll
        for (int c = 0; c < 3; ++c)
            atomicAdd(&ls[id * 3 + c], y[(long)(b * 3 + c) * P + p]);
    }
    __syncthreads();
    for (int i = threadIdx.x; i < 96; i += 256) atomicAdd(&sums[b * 96 + i], ls[i]);
    for (int i = threadIdx.x; i < 32; i += 256) atomicAdd(&cnt[b * 32 + i], lc[i]);
}

__launch_bounds__(256)
__global__ void pool_scatter(const int* __restrict__ ids, const float* __restrict__ sums,
                             const float* __restrict__ cnt, float* out) {
    const int b = blockIdx.y;
    const int P = 512 * 512;
    const int p = blockIdx.x * 256 + threadIdx.x;
    if (p >= P) return;
    const int id = ids[(long)b * P + p] & 31;
    float c = cnt[b * 32 + id];
    c = c > 1.0f ? c : 1.0f;
#pragma unroll
    for (int ch = 0; ch < 3; ++ch)
        out[(long)(b * 3 + ch) * P + p] = sums[(b * 32 + id) * 3 + ch] / c;
}

// ---------------------------------------------------------------------------
static inline int ilog2i(int x) { int l = 0; while ((1 << l) < x) ++l; return l; }

template <int KSZ, bool TRANS, bool REFLECT, bool TANH>
static void launch_conv(const _Float16* in, const float* w, const float* bias,
                        float* out, _Float16* wpack,
                        int Cin, int Hin, int Cout, int Hout,
                        int stride, int pad, hipStream_t stream) {
    const int KK = KSZ * KSZ;
    const int Ktot = Cin * KK;
    const int NC = (Ktot + 31) / 32;
    const int ntiles = (Cout + 15) / 16;
    hipLaunchKernelGGL(pack_weights, dim3(NC, ntiles), dim3(32), 0, stream,
                       w, wpack, Cin, Cout, KSZ, TRANS ? 1 : 0);
    dim3 g((Hout * Hout) / 64, (ntiles + 3) / 4, 8);
    hipLaunchKernelGGL((conv_wmma<KSZ, TRANS, REFLECT, TANH>), g, dim3(128), 0, stream,
                       in, wpack, bias, out,
                       Cin, ilog2i(Hin), Cout, ilog2i(Hout), stride, pad, NC);
}

extern "C" void kernel_launch(void* const* d_in, const int* in_sizes, int n_in,
                              void* d_out, int out_size, void* d_ws, size_t ws_size,
                              hipStream_t stream) {
    (void)in_sizes; (void)n_in; (void)out_size; (void)ws_size;

    const float* x    = (const float*)d_in[0];
    const int*   imap = (const int*)d_in[1];
    const float* w_in = (const float*)d_in[2];
    const float* b_in = (const float*)d_in[3];
    const float *wd[4], *bd[4], *wu[4], *bu[4];
    for (int i = 0; i < 4; ++i) { wd[i] = (const float*)d_in[4 + 2 * i];  bd[i] = (const float*)d_in[5 + 2 * i]; }
    for (int i = 0; i < 4; ++i) { wu[i] = (const float*)d_in[12 + 2 * i]; bu[i] = (const float*)d_in[13 + 2 * i]; }
    const float* w_out = (const float*)d_in[20];
    const float* b_out = (const float*)d_in[21];
    float* out = (float*)d_out;

    // Workspace (float units):
    //   raw  : 33,554,432 f32  (per-layer raw conv output, reused every layer)
    //   hA/hB: 33,554,432 f16 each (normalized activations, ping-pong)
    //   stats + pooling cells + packed weights
    float* ws   = (float*)d_ws;
    float* raw  = ws;
    _Float16* hA = (_Float16*)(ws + 33554432L);
    _Float16* hB = (_Float16*)(ws + 50331648L);
    float* mean = ws + 67108864L;
    float* inv  = mean + 2048;
    float* sums = inv + 2048;
    float* cnt  = sums + 768;
    _Float16* wpack = (_Float16*)(cnt + 256);

    auto stats = [&](const float* buf, int C, int HW) {
        hipLaunchKernelGGL(inorm_stats, dim3(8 * C), dim3(256), 0, stream, buf, mean, inv, HW);
    };
    auto normf16 = [&](const float* src, _Float16* dst, int C, int HW, int relu,
                       bool identity) {
        dim3 g((HW / 4 + 255) / 256, 8 * C);
        hipLaunchKernelGGL(norm_to_f16, g, dim3(256), 0, stream,
                           src, identity ? nullptr : mean, identity ? nullptr : inv,
                           dst, HW, relu);
    };

    // input f32 -> f16 (identity)
    normf16(x, hA, 3, 512 * 512, 0, true);

    // conv_in: 7x7 reflect(3)
    launch_conv<7, false, true, false>(hA, w_in, b_in, raw, wpack, 3, 512, 16, 512, 1, 3, stream);
    stats(raw, 16, 512 * 512);
    normf16(raw, hB, 16, 512 * 512, 1, false);
    // down path: 3x3 s2 p1
    launch_conv<3, false, false, false>(hB, wd[0], bd[0], raw, wpack, 16, 512, 32, 256, 2, 1, stream);
    stats(raw, 32, 256 * 256);
    normf16(raw, hA, 32, 256 * 256, 1, false);
    launch_conv<3, false, false, false>(hA, wd[1], bd[1], raw, wpack, 32, 256, 64, 128, 2, 1, stream);
    stats(raw, 64, 128 * 128);
    normf16(raw, hB, 64, 128 * 128, 1, false);
    launch_conv<3, false, false, false>(hB, wd[2], bd[2], raw, wpack, 64, 128, 128, 64, 2, 1, stream);
    stats(raw, 128, 64 * 64);
    normf16(raw, hA, 128, 64 * 64, 1, false);
    launch_conv<3, false, false, false>(hA, wd[3], bd[3], raw, wpack, 128, 64, 256, 32, 2, 1, stream);
    stats(raw, 256, 32 * 32);
    normf16(raw, hB, 256, 32 * 32, 1, false);
    // up path: transposed 3x3
    launch_conv<3, true, false, false>(hB, wu[0], bu[0], raw, wpack, 256, 32, 128, 64, 1, 0, stream);
    stats(raw, 128, 64 * 64);
    normf16(raw, hA, 128, 64 * 64, 1, false);
    launch_conv<3, true, false, false>(hA, wu[1], bu[1], raw, wpack, 128, 64, 64, 128, 1, 0, stream);
    stats(raw, 64, 128 * 128);
    normf16(raw, hB, 64, 128 * 128, 1, false);
    launch_conv<3, true, false, false>(hB, wu[2], bu[2], raw, wpack, 64, 128, 32, 256, 1, 0, stream);
    stats(raw, 32, 256 * 256);
    normf16(raw, hA, 32, 256 * 256, 1, false);
    launch_conv<3, true, false, false>(hA, wu[3], bu[3], raw, wpack, 32, 256, 16, 512, 1, 0, stream);
    stats(raw, 16, 512 * 512);
    normf16(raw, hB, 16, 512 * 512, 1, false);
    // conv_out: 7x7 reflect(3) + tanh -> d_out
    launch_conv<7, false, true, true>(hB, w_out, b_out, out, wpack, 16, 512, 3, 512, 1, 3, stream);

    // instance-wise mean pooling (reads d_out, then overwrites d_out)
    hipLaunchKernelGGL(pool_zero, dim3(4), dim3(256), 0, stream, sums);
    dim3 pg((512 * 512 + 255) / 256, 8);
    hipLaunchKernelGGL(pool_accum, pg, dim3(256), 0, stream, out, imap, sums, cnt);
    hipLaunchKernelGGL(pool_scatter, pg, dim3(256), 0, stream, imap, sums, cnt, out);
}
